// DimeNet_49984829391023
// MI455X (gfx1250) — compile-verified, matrix operand order checked by reading, not run on hardware
//
#include <hip/hip_runtime.h>

// ---------------------------------------------------------------------------
// DimeNet-style GNN forward for MI455X (gfx1250), wave32 + WMMA bf16.
// v3: per-triplet tail row (rbf|cbf|pad) precomputed in bf16 -> the triplet
// kernel's A-fragments are three unconditional 32B vector loads per lane
// (no divergent scalar gathers), and the rbf[j_idx] gather is hoisted out of
// the layer loop. fp32 accumulation everywhere (WMMA C/D + global atomics).
// ---------------------------------------------------------------------------

typedef __attribute__((ext_vector_type(16))) __bf16 v16bf;
typedef __attribute__((ext_vector_type(8)))  float  v8f;

#define N_NODES 16000
#define N_EDGES 256000
#define N_TRIP  640000
#define N_GRAPH 128
#define F_IN    64
#define HDIM    64
#define OUT_DIM 32
#define NRBF    6
#define NCBF    6
#define NLAYER  3
#define K1PAD   96     // 76 features padded to 3 K-chunks of 32
#define K2      128
#define K3      64
#define TAILW   32     // tail row: 6 rbf | 6 cbf | 20 zeros

__device__ __forceinline__ unsigned short f2bfbits(float f) {
  union { float f; unsigned u; } c; c.f = f;
  unsigned r = c.u + 0x7FFFu + ((c.u >> 16) & 1u);   // round-to-nearest-even
  return (unsigned short)(r >> 16);
}
__device__ __forceinline__ __bf16 bits2bf(unsigned short s) {
  union { unsigned short s; __bf16 b; } c; c.s = s; return c.b;
}
__device__ __forceinline__ __bf16 f2bf(float f) { return bits2bf(f2bfbits(f)); }
__device__ __forceinline__ float bf2f(unsigned short s) {
  union { unsigned u; float f; } c; c.u = (unsigned)s << 16; return c.f;
}

__device__ __forceinline__ v8f v8f_zero() {
  v8f z = {0.f, 0.f, 0.f, 0.f, 0.f, 0.f, 0.f, 0.f};
  return z;
}

// ---------------------------------------------------------------------------
__global__ void zero_f32(float* __restrict__ p, long long n) {
  long long i = (long long)blockIdx.x * blockDim.x + threadIdx.x;
  long long stride = (long long)gridDim.x * blockDim.x;
  for (; i < n; i += stride) p[i] = 0.f;
}

// fp32 -> bf16 copy of node features x.
__global__ void convert_x(const float* __restrict__ x, unsigned short* __restrict__ xb) {
  long long i = (long long)blockIdx.x * blockDim.x + threadIdx.x;
  long long stride = (long long)gridDim.x * blockDim.x;
  for (long long k = i; k < (long long)N_NODES * F_IN; k += stride) xb[k] = f2bfbits(x[k]);
}

// Layer-invariant per-triplet tail row: rbf[j_idx[t]] | cbf[t] | zero pad.
// One thread per triplet writes a 64B contiguous row (block writes are fully
// contiguous across threads).
__global__ void build_tail(const float* __restrict__ rbf, const float* __restrict__ cbf,
                           const int* __restrict__ j_idx, unsigned short* __restrict__ tail) {
  int t = blockIdx.x * blockDim.x + threadIdx.x;
  if (t >= N_TRIP) return;
  const float* rr = rbf + (long long)j_idx[t] * NRBF;
  const float* cc = cbf + (long long)t * NCBF;
  unsigned short* row = tail + (long long)t * TAILW;
#pragma unroll
  for (int e = 0; e < NRBF; ++e)        row[e] = f2bfbits(rr[e]);
#pragma unroll
  for (int e = 0; e < NCBF; ++e)        row[NRBF + e] = f2bfbits(cc[e]);
#pragma unroll
  for (int e = NRBF + NCBF; e < TAILW; ++e) row[e] = 0;
}

// Convert all layer weights fp32 -> bf16, transposed to [layer][n][k] so each
// lane's B-fragment (16 consecutive K values of one column) is contiguous.
__global__ void convert_weights(const float* __restrict__ W1, const float* __restrict__ W2,
                                const float* __restrict__ Wn1, const float* __restrict__ Wn2,
                                unsigned short* __restrict__ W1t, unsigned short* __restrict__ W2t,
                                unsigned short* __restrict__ Wn1t, unsigned short* __restrict__ Wn2t) {
  int idx = blockIdx.x * blockDim.x + threadIdx.x;
  if (idx < NLAYER * HDIM * K1PAD) {
    int l = idx / (HDIM * K1PAD), r = idx % (HDIM * K1PAD);
    int n = r / K1PAD, k = r % K1PAD;
    float v = (k < F_IN + NRBF + NCBF) ? W1[((long long)l * (F_IN + NRBF + NCBF) + k) * HDIM + n] : 0.f;
    W1t[idx] = f2bfbits(v);
  }
  if (idx < NLAYER * HDIM * K2) {
    int l = idx / (HDIM * K2), r = idx % (HDIM * K2);
    int n = r / K2, k = r % K2;
    W2t[idx]  = f2bfbits(W2 [((long long)l * K2 + k) * HDIM + n]);
    Wn1t[idx] = f2bfbits(Wn1[((long long)l * K2 + k) * HDIM + n]);
  }
  if (idx < NLAYER * HDIM * K3) {
    int l = idx / (HDIM * K3), r = idx % (HDIM * K3);
    int n = r / K3, k = r % K3;
    Wn2t[idx] = f2bfbits(Wn2[((long long)l * K3 + k) * HDIM + n]);
  }
}

// ---------------------------------------------------------------------------
// Triplet MLP: em = relu(concat(h[k_idx], rbf[j_idx], cbf) @ W1 + b1)
// fused with scatter-add onto agg_e[j_idx].  One wave = 16 triplets x 64 cols.
__global__ __launch_bounds__(256) void triplet_mlp(
    const unsigned short* __restrict__ hin, const unsigned short* __restrict__ tail,
    const unsigned short* __restrict__ W1t, const float* __restrict__ b1,
    const int* __restrict__ k_idx, const int* __restrict__ j_idx,
    float* __restrict__ agg_e) {
  __shared__ __align__(32) unsigned short wlds[HDIM * K1PAD];   // 12 KB, [n][k]
  for (int i = threadIdx.x; i < HDIM * K1PAD; i += blockDim.x) wlds[i] = W1t[i];
  __syncthreads();

  const int lane = threadIdx.x & 31;
  const int wave = threadIdx.x >> 5;
  const int t0 = (blockIdx.x * 8 + wave) * 16;
  const int m    = lane & 15;
  const int half = lane >> 4;

  // A-fragment source row for this lane (16-bit A layout: M = lane%16)
  const int rA = t0 + m;
  const unsigned short* hrow = hin  + (long long)k_idx[rA] * F_IN;
  const unsigned short* trow = tail + (long long)rA * TAILW;

  v8f acc[4];
#pragma unroll
  for (int nt = 0; nt < 4; ++nt) acc[nt] = v8f_zero();

  // Three K-chunks of 32: h[0:32), h[32:64), tail (rbf|cbf|pad).
  // Every A slice is one unconditional 32B contiguous vector load.
#pragma unroll
  for (int kc = 0; kc < 3; ++kc) {
    const int kbase = kc * 32 + half * 16;
    const unsigned short* arow = (kc < 2) ? &hrow[kbase] : &trow[half * 16];
    v16bf a;
    __builtin_memcpy(&a, arow, sizeof(a));
#pragma unroll
    for (int nt = 0; nt < 4; ++nt) {
      v16bf b;
      __builtin_memcpy(&b, &wlds[(nt * 16 + m) * K1PAD + kbase], sizeof(b));
      acc[nt] = __builtin_amdgcn_wmma_f32_16x16x32_bf16(false, a, false, b,
                                                        (short)0, acc[nt], false, false);
    }
  }

  // Epilogue: D layout -> lane holds rows t0+8*half+v, col nt*16+m.
  const int mrow = half * 8;
  int jdst[8];
#pragma unroll
  for (int v = 0; v < 8; ++v) jdst[v] = j_idx[t0 + mrow + v];
#pragma unroll
  for (int nt = 0; nt < 4; ++nt) {
    const int c = nt * 16 + m;
    const float bias = b1[c];
#pragma unroll
    for (int v = 0; v < 8; ++v) {
      float val = fmaxf(acc[nt][v] + bias, 0.f);
      unsafeAtomicAdd(&agg_e[(long long)jdst[v] * HDIM + c], val);
    }
  }
}

// ---------------------------------------------------------------------------
// Edge MLP: fi = relu(concat(h[src], agg_e[dst]) @ W2 + b2), fused with
// scatter-add onto aggr[dst].
__global__ __launch_bounds__(256) void edge_mlp(
    const unsigned short* __restrict__ hin, const float* __restrict__ agg_e,
    const unsigned short* __restrict__ W2t, const float* __restrict__ b2,
    const int* __restrict__ src, const int* __restrict__ dst,
    float* __restrict__ aggr) {
  __shared__ __align__(32) unsigned short wlds[HDIM * K2];      // 16 KB
  for (int i = threadIdx.x; i < HDIM * K2; i += blockDim.x) wlds[i] = W2t[i];
  __syncthreads();

  const int lane = threadIdx.x & 31;
  const int wave = threadIdx.x >> 5;
  const int t0 = (blockIdx.x * 8 + wave) * 16;
  const int m    = lane & 15;
  const int half = lane >> 4;

  const int rA = t0 + m;
  const unsigned short* hrow = hin + (long long)src[rA] * F_IN;
  const float* arow = agg_e + (long long)dst[rA] * HDIM;   // agg_e[dst] per reference

  v8f acc[4];
#pragma unroll
  for (int nt = 0; nt < 4; ++nt) acc[nt] = v8f_zero();

#pragma unroll
  for (int kc = 0; kc < 4; ++kc) {
    const int kbase = kc * 32 + half * 16;
    v16bf a;
    if (kc < 2) {                          // h half: direct bf16 vector load
      __builtin_memcpy(&a, &hrow[kbase], sizeof(a));
    } else {                               // agg_e half: fp32 -> bf16 on load
#pragma unroll
      for (int e = 0; e < 16; ++e) a[e] = f2bf(arow[kbase - F_IN + e]);
    }
#pragma unroll
    for (int nt = 0; nt < 4; ++nt) {
      v16bf b;
      __builtin_memcpy(&b, &wlds[(nt * 16 + m) * K2 + kbase], sizeof(b));
      acc[nt] = __builtin_amdgcn_wmma_f32_16x16x32_bf16(false, a, false, b,
                                                        (short)0, acc[nt], false, false);
    }
  }

  const int mrow = half * 8;
  int ndst[8];
#pragma unroll
  for (int v = 0; v < 8; ++v) ndst[v] = dst[t0 + mrow + v];
#pragma unroll
  for (int nt = 0; nt < 4; ++nt) {
    const int c = nt * 16 + m;
    const float bias = b2[c];
#pragma unroll
    for (int v = 0; v < 8; ++v) {
      float val = fmaxf(acc[nt][v] + bias, 0.f);
      unsafeAtomicAdd(&aggr[(long long)ndst[v] * HDIM + c], val);
    }
  }
}

// ---------------------------------------------------------------------------
// Node update: z = relu(concat(h, aggr) @ Wn1 + bn1);  hout = z @ Wn2 + bn2.
// Both GEMMs fused; z bounces through a per-wave LDS tile (D layout -> A layout).
// hout written directly as bf16 for the next layer.
__global__ __launch_bounds__(256) void node_mlp(
    const unsigned short* __restrict__ hin, const float* __restrict__ aggr,
    const unsigned short* __restrict__ Wn1t, const float* __restrict__ bn1,
    const unsigned short* __restrict__ Wn2t, const float* __restrict__ bn2,
    unsigned short* __restrict__ hout) {
  __shared__ __align__(32) unsigned short w1lds[HDIM * K2];       // 16 KB
  __shared__ __align__(32) unsigned short w2lds[HDIM * K3];       //  8 KB
  __shared__ __align__(32) unsigned short ztile[8][16 * HDIM];    // 16 KB
  for (int i = threadIdx.x; i < HDIM * K2; i += blockDim.x) w1lds[i] = Wn1t[i];
  for (int i = threadIdx.x; i < HDIM * K3; i += blockDim.x) w2lds[i] = Wn2t[i];
  __syncthreads();

  const int lane = threadIdx.x & 31;
  const int wave = threadIdx.x >> 5;
  const int t0 = (blockIdx.x * 8 + wave) * 16;
  const int m    = lane & 15;
  const int half = lane >> 4;
  const int mrow = half * 8;

  const int rA = t0 + m;
  const unsigned short* hrow = hin + (long long)rA * F_IN;
  const float* arow = aggr + (long long)rA * HDIM;

  // ---- GEMM 1: K = 128 (concat h | aggr) ----
  v8f acc[4];
#pragma unroll
  for (int nt = 0; nt < 4; ++nt) acc[nt] = v8f_zero();

#pragma unroll
  for (int kc = 0; kc < 4; ++kc) {
    const int kbase = kc * 32 + half * 16;
    v16bf a;
    if (kc < 2) {
      __builtin_memcpy(&a, &hrow[kbase], sizeof(a));
    } else {
#pragma unroll
      for (int e = 0; e < 16; ++e) a[e] = f2bf(arow[kbase - F_IN + e]);
    }
#pragma unroll
    for (int nt = 0; nt < 4; ++nt) {
      v16bf b;
      __builtin_memcpy(&b, &w1lds[(nt * 16 + m) * K2 + kbase], sizeof(b));
      acc[nt] = __builtin_amdgcn_wmma_f32_16x16x32_bf16(false, a, false, b,
                                                        (short)0, acc[nt], false, false);
    }
  }

  // bias + relu, park z tile in LDS as bf16 (row-major [16][64])
#pragma unroll
  for (int nt = 0; nt < 4; ++nt) {
    const int c = nt * 16 + m;
    const float bias = bn1[c];
#pragma unroll
    for (int v = 0; v < 8; ++v)
      ztile[wave][(mrow + v) * HDIM + c] = f2bfbits(fmaxf(acc[nt][v] + bias, 0.f));
  }
  // same-wave LDS store->load: drain DScnt and fence compiler reordering
  asm volatile("s_wait_dscnt 0x0" ::: "memory");

  // ---- GEMM 2: K = 64 ----
  v8f acc2[4];
#pragma unroll
  for (int nt = 0; nt < 4; ++nt) acc2[nt] = v8f_zero();

#pragma unroll
  for (int kc = 0; kc < 2; ++kc) {
    const int kbase = kc * 32 + half * 16;
    v16bf a;
    __builtin_memcpy(&a, &ztile[wave][m * HDIM + kbase], sizeof(a));
#pragma unroll
    for (int nt = 0; nt < 4; ++nt) {
      v16bf b;
      __builtin_memcpy(&b, &w2lds[(nt * 16 + m) * K3 + kbase], sizeof(b));
      acc2[nt] = __builtin_amdgcn_wmma_f32_16x16x32_bf16(false, a, false, b,
                                                         (short)0, acc2[nt], false, false);
    }
  }

#pragma unroll
  for (int nt = 0; nt < 4; ++nt) {
    const int c = nt * 16 + m;
    const float bias = bn2[c];
#pragma unroll
    for (int v = 0; v < 8; ++v)
      hout[(long long)(t0 + mrow + v) * HDIM + c] = f2bfbits(acc2[nt][v] + bias);
  }
}

// ---------------------------------------------------------------------------
__global__ void pool_accum(const unsigned short* __restrict__ h, const int* __restrict__ batch,
                           float* __restrict__ pooled, float* __restrict__ cnt) {
  int idx = blockIdx.x * blockDim.x + threadIdx.x;
  if (idx >= N_NODES * HDIM) return;
  const int n = idx >> 6, c = idx & 63;
  const int g = batch[n];
  unsafeAtomicAdd(&pooled[g * HDIM + c], bf2f(h[idx]));
  if (c == 0) unsafeAtomicAdd(&cnt[g], 1.f);
}

__global__ void head_mlp(const float* __restrict__ pooled, const float* __restrict__ cnt,
                         const float* __restrict__ Wo1, const float* __restrict__ bo1,
                         const float* __restrict__ Wo2, const float* __restrict__ bo2,
                         float* __restrict__ out) {
  int g = blockIdx.x * blockDim.x + threadIdx.x;
  if (g >= N_GRAPH) return;
  const float inv = 1.f / fmaxf(cnt[g], 1.f);
  float p[HDIM];
  for (int k = 0; k < HDIM; ++k) p[k] = fmaxf(pooled[g * HDIM + k] * inv, 0.f);
  float y[HDIM];
  for (int j = 0; j < HDIM; ++j) {
    float s = bo1[j];
    for (int k = 0; k < HDIM; ++k) s += p[k] * Wo1[k * HDIM + j];
    y[j] = fmaxf(s, 0.f);
  }
  for (int o = 0; o < OUT_DIM; ++o) {
    float s = bo2[o];
    for (int j = 0; j < HDIM; ++j) s += y[j] * Wo2[j * OUT_DIM + o];
    out[g * OUT_DIM + o] = s;
  }
}

// ---------------------------------------------------------------------------
extern "C" void kernel_launch(void* const* d_in, const int* in_sizes, int n_in,
                              void* d_out, int out_size, void* d_ws, size_t ws_size,
                              hipStream_t stream) {
  const float* x   = (const float*)d_in[0];
  const float* rbf = (const float*)d_in[1];
  const float* cbf = (const float*)d_in[2];
  const float* W1  = (const float*)d_in[3];
  const float* b1  = (const float*)d_in[4];
  const float* W2  = (const float*)d_in[5];
  const float* b2  = (const float*)d_in[6];
  const float* Wn1 = (const float*)d_in[7];
  const float* bn1 = (const float*)d_in[8];
  const float* Wn2 = (const float*)d_in[9];
  const float* bn2 = (const float*)d_in[10];
  const float* Wo1 = (const float*)d_in[11];
  const float* bo1 = (const float*)d_in[12];
  const float* Wo2 = (const float*)d_in[13];
  const float* bo2 = (const float*)d_in[14];
  const int* edge_index = (const int*)d_in[15];
  const int* k_idx = (const int*)d_in[16];
  const int* j_idx = (const int*)d_in[17];
  const int* batch = (const int*)d_in[18];
  const int* src = edge_index;             // edge_index[0]
  const int* dst = edge_index + N_EDGES;   // edge_index[1]

  char* ws = (char*)d_ws;
  size_t off = 0;
  auto alloc = [&](size_t bytes) -> void* {
    void* p = ws + off;
    off += (bytes + 255) & ~(size_t)255;
    return p;
  };
  float* agg_e  = (float*)alloc((size_t)N_EDGES * HDIM * 4);   // 65.5 MB (L2-resident)
  float* aggr   = (float*)alloc((size_t)N_NODES * HDIM * 4);
  float* pooled = (float*)alloc((size_t)N_GRAPH * HDIM * 4);
  float* cnt    = (float*)alloc((size_t)N_GRAPH * 4);
  unsigned short* xb   = (unsigned short*)alloc((size_t)N_NODES * F_IN * 2);
  unsigned short* hb0  = (unsigned short*)alloc((size_t)N_NODES * HDIM * 2);
  unsigned short* hb1  = (unsigned short*)alloc((size_t)N_NODES * HDIM * 2);
  unsigned short* tail = (unsigned short*)alloc((size_t)N_TRIP * TAILW * 2);  // 41 MB
  unsigned short* W1t  = (unsigned short*)alloc((size_t)NLAYER * HDIM * K1PAD * 2);
  unsigned short* W2t  = (unsigned short*)alloc((size_t)NLAYER * HDIM * K2 * 2);
  unsigned short* Wn1t = (unsigned short*)alloc((size_t)NLAYER * HDIM * K2 * 2);
  unsigned short* Wn2t = (unsigned short*)alloc((size_t)NLAYER * HDIM * K3 * 2);

  convert_x<<<1024, 256, 0, stream>>>(x, xb);
  build_tail<<<(N_TRIP + 255) / 256, 256, 0, stream>>>(rbf, cbf, j_idx, tail);
  convert_weights<<<(NLAYER * HDIM * K2 + 255) / 256, 256, 0, stream>>>(
      W1, W2, Wn1, Wn2, W1t, W2t, Wn1t, Wn2t);

  const unsigned short* hin = xb;
  unsigned short* hbuf[2] = {hb0, hb1};
  for (int l = 0; l < NLAYER; ++l) {
    zero_f32<<<2048, 256, 0, stream>>>(agg_e, (long long)N_EDGES * HDIM);
    zero_f32<<<512, 256, 0, stream>>>(aggr, (long long)N_NODES * HDIM);
    triplet_mlp<<<N_TRIP / 128, 256, 0, stream>>>(
        hin, tail, W1t + (size_t)l * HDIM * K1PAD, b1 + l * HDIM, k_idx, j_idx, agg_e);
    edge_mlp<<<N_EDGES / 128, 256, 0, stream>>>(
        hin, agg_e, W2t + (size_t)l * HDIM * K2, b2 + l * HDIM, src, dst, aggr);
    unsigned short* hout = hbuf[l & 1];
    node_mlp<<<N_NODES / 128, 256, 0, stream>>>(
        hin, aggr, Wn1t + (size_t)l * HDIM * K2, bn1 + l * HDIM,
        Wn2t + (size_t)l * HDIM * K3, bn2 + l * HDIM, hout);
    hin = hout;
  }

  zero_f32<<<16, 256, 0, stream>>>(pooled, (long long)N_GRAPH * HDIM);
  zero_f32<<<1, 128, 0, stream>>>(cnt, N_GRAPH);
  pool_accum<<<(N_NODES * HDIM + 255) / 256, 256, 0, stream>>>(hin, batch, pooled, cnt);
  head_mlp<<<1, 128, 0, stream>>>(pooled, cnt, Wo1, bo1, Wo2, bo2, (float*)d_out);
}
